// PaiNN_35175782154416
// MI455X (gfx1250) — matrix operand order repaired and checked by reading, not compiled, not run on hardware
//
#include <hip/hip_runtime.h>
#include <math.h>

#define Hdim 128
#define H3   384
#define PI_  3.14159265358979323846f

typedef __attribute__((ext_vector_type(16))) __bf16 v16bf;
typedef __attribute__((ext_vector_type(8)))  __bf16 v8bf;
typedef __attribute__((ext_vector_type(8)))  float  v8f;

#define WMMA_BF16(a, b, c) \
    __builtin_amdgcn_wmma_f32_16x16x32_bf16(false, (a), false, (b), (short)0, (c), false, false)

__device__ __forceinline__ float clipf(float x) { return fminf(fmaxf(x, -1e4f), 1e4f); }
__device__ __forceinline__ float siluf(float x) { return x / (1.0f + __expf(-x)); }

// A fragment: 16x32 (MxK) bf16, row-major, leading dim lda (fragment chunks 16B aligned).
// Lanes 0-15: row=lane, K {0..15}; lanes 16-31: row=lane-16, K {16..31}.
__device__ __forceinline__ v16bf load_A(const __bf16* A, int lda, int lane) {
    int row  = lane & 15;
    int koff = (lane >> 4) << 4;
    const __bf16* p = A + row * lda + koff;
    v8bf lo = *(const v8bf*)p;
    v8bf hi = *(const v8bf*)(p + 8);
    return __builtin_shufflevector(lo, hi, 0, 1, 2, 3, 4, 5, 6, 7,
                                           8, 9, 10, 11, 12, 13, 14, 15);
}

// B fragment from pre-packed weights: tile-major [tile][lane][16], 32B/lane contiguous.
__device__ __forceinline__ v16bf load_Bp(const __bf16* Bp, long tile, int lane) {
    return *(const v16bf*)(Bp + ((size_t)tile * 32 + lane) * 16);
}

// ---------------------------------------------------------------------------
// Fused 2-layer MLP: out[M,384] = silu(A[M,K1] @ W1[K1,128] + b1) @ W2[128,384] + b2
// W1p/W2p are B-fragment packed. One wave per 16-row M-tile.
// ---------------------------------------------------------------------------
template <int K1>
__global__ void __launch_bounds__(32) fused_mlp_kernel(
    const __bf16* __restrict__ A, const __bf16* __restrict__ W1p,
    const float* __restrict__ b1, const __bf16* __restrict__ W2p,
    const float* __restrict__ b2, float* __restrict__ out)
{
    __shared__ __bf16 hbuf[16 * Hdim];
    const int lane  = threadIdx.x;
    const int col   = lane & 15;
    const int rbase = (lane >> 4) << 3;
    const size_t m0 = (size_t)blockIdx.x * 16;
    const __bf16* Abase = A + m0 * K1;

#pragma unroll
    for (int nt = 0; nt < Hdim / 16; ++nt) {
        v8f acc;
        float bias = b1[nt * 16 + col];
#pragma unroll
        for (int r = 0; r < 8; ++r) acc[r] = bias;
#pragma unroll
        for (int kt = 0; kt < K1 / 32; ++kt) {
            v16bf a = load_A(Abase + kt * 32, K1, lane);
            v16bf b = load_Bp(W1p, (long)kt * (Hdim / 16) + nt, lane);
            acc = WMMA_BF16(a, b, acc);
        }
#pragma unroll
        for (int r = 0; r < 8; ++r)
            hbuf[(rbase + r) * Hdim + nt * 16 + col] = (__bf16)siluf(acc[r]);
    }
    __syncthreads();
    for (int nt = 0; nt < H3 / 16; ++nt) {
        v8f acc;
        float bias = b2[nt * 16 + col];
#pragma unroll
        for (int r = 0; r < 8; ++r) acc[r] = bias;
#pragma unroll
        for (int kt = 0; kt < Hdim / 32; ++kt) {
            v16bf a = load_A(hbuf + kt * 32, Hdim, lane);
            v16bf b = load_Bp(W2p, (long)kt * (H3 / 16) + nt, lane);
            acc = WMMA_BF16(a, b, acc);
        }
        float* o = out + m0 * H3 + nt * 16;
#pragma unroll
        for (int r = 0; r < 8; ++r) o[(size_t)(rbase + r) * H3 + col] = acc[r];
    }
}

// ---------------------------------------------------------------------------
// v @ vW: out[M,256] = A[M,128] @ Wp[128,256] (packed)   (M = 3*N rows)
// ---------------------------------------------------------------------------
__global__ void __launch_bounds__(32) gemm_vw_kernel(
    const __bf16* __restrict__ A, const __bf16* __restrict__ Wp,
    float* __restrict__ out)
{
    const int lane  = threadIdx.x;
    const int col   = lane & 15;
    const int rbase = (lane >> 4) << 3;
    const size_t m0 = (size_t)blockIdx.x * 16;
    for (int nt = 0; nt < 16; ++nt) {
        v8f acc;
#pragma unroll
        for (int r = 0; r < 8; ++r) acc[r] = 0.f;
#pragma unroll
        for (int kt = 0; kt < 4; ++kt) {
            v16bf a = load_A(A + m0 * Hdim + kt * 32, Hdim, lane);
            v16bf b = load_Bp(Wp, (long)kt * 16 + nt, lane);
            acc = WMMA_BF16(a, b, acc);
        }
        float* o = out + m0 * 256 + nt * 16;
#pragma unroll
        for (int r = 0; r < 8; ++r) o[(size_t)(rbase + r) * 256 + col] = acc[r];
    }
}

// ---------------------------------------------------------------------------
// Edge kernel: per 16-edge tile, WMMA-compute the layer's filters on the fly
// (phi*cut padded to K=32, bias folded into row 20 of wPad), then fused
// message formation + atomic scatter-add over senders.
// ---------------------------------------------------------------------------
__global__ void __launch_bounds__(32) edge_kernel(
    const __bf16* __restrict__ phiB,   // [E,32]
    const __bf16* __restrict__ wPadP,  // packed [32,1152] -> 72 tiles
    const float*  __restrict__ xF,     // [N,384]
    const float*  __restrict__ vF,     // [N,3,128]
    const float*  __restrict__ dirF,   // [E,3]
    const int*    __restrict__ senders,
    const int*    __restrict__ receivers,
    float* __restrict__ dsAcc,         // [N,128]
    float* __restrict__ dvAcc,         // [N,3,128]
    int layer)
{
    __shared__ float filt[16 * H3];
    __shared__ int   sh_snd[16];
    __shared__ int   sh_rcv[16];
    __shared__ float sh_dir[16][3];
    const int lane  = threadIdx.x;
    const int col   = lane & 15;
    const int rbase = (lane >> 4) << 3;
    const size_t e0 = (size_t)blockIdx.x * 16;

    if (lane < 16) {
        size_t e = e0 + lane;
        sh_rcv[lane]    = receivers[e];
        sh_snd[lane]    = senders[e];
        sh_dir[lane][0] = dirF[e * 3 + 0];
        sh_dir[lane][1] = dirF[e * 3 + 1];
        sh_dir[lane][2] = dirF[e * 3 + 2];
    }

    v16bf a = load_A(phiB + e0 * 32, 32, lane);
    for (int nt = 0; nt < 24; ++nt) {
        v8f acc;
#pragma unroll
        for (int r = 0; r < 8; ++r) acc[r] = 0.f;
        v16bf b = load_Bp(wPadP, (long)layer * 24 + nt, lane);
        acc = WMMA_BF16(a, b, acc);
#pragma unroll
        for (int r = 0; r < 8; ++r)
            filt[(rbase + r) * H3 + nt * 16 + col] = acc[r];
    }
    __syncthreads();

    for (int i = 0; i < 64; ++i) {              // 16 edges * 128 features / 32 lanes
        int idx = i * 32 + lane;
        int el  = idx >> 7;
        int h   = idx & 127;
        int rcv = sh_rcv[el];
        int snd = sh_snd[el];
        const float* xr = xF + (size_t)rcv * H3;
        float f0 = filt[el * H3 + h];
        float f1 = filt[el * H3 + 128 + h];
        float f2 = filt[el * H3 + 256 + h];
        float ds  = f0 * xr[h];
        float dv1 = f1 * xr[128 + h];
        float dv2 = f2 * xr[256 + h];
        float dx = sh_dir[el][0], dy = sh_dir[el][1], dz = sh_dir[el][2];
        const float* vr = vF + (size_t)rcv * H3;
        unsafeAtomicAdd(dsAcc + (size_t)snd * Hdim + h, ds);
        unsafeAtomicAdd(dvAcc + (size_t)snd * H3 + h,        dv1 * dx + dv2 * vr[h]);
        unsafeAtomicAdd(dvAcc + (size_t)snd * H3 + 128 + h,  dv1 * dy + dv2 * vr[128 + h]);
        unsafeAtomicAdd(dvAcc + (size_t)snd * H3 + 256 + h,  dv1 * dz + dv2 * vr[256 + h]);
    }
}

// ---------------------------------------------------------------------------
// Prep kernels
// ---------------------------------------------------------------------------
__global__ void zero_f32(float* __restrict__ p, long n) {
    long i = (long)blockIdx.x * blockDim.x + threadIdx.x;
    if (i < n) p[i] = 0.f;
}

// Pack one weight matrix W[K, Ncols] (f32 row-major) into WMMA B-fragment order:
// out[((kt*NT + nt)*32 + lane)*16 + j] = W[(kt*32 + (lane>>4)*16 + j) * Ncols + nt*16 + (lane&15)]
__global__ void pack_B(const float* __restrict__ W, __bf16* __restrict__ out,
                       int Ncols, long total) {
    long i = (long)blockIdx.x * blockDim.x + threadIdx.x;
    if (i >= total) return;
    int  j    = (int)(i & 15);
    int  lane = (int)((i >> 4) & 31);
    long tile = i >> 9;
    int  NT   = Ncols >> 4;
    int  kt   = (int)(tile / NT);
    int  nt   = (int)(tile % NT);
    int  c    = nt * 16 + (lane & 15);
    int  k    = kt * 32 + ((lane >> 4) << 4) + j;
    out[i] = (__bf16)W[(size_t)k * Ncols + c];
}

// Build + pack the padded filter weight [32 x 1152]: rows 0..19 = filt_W,
// row 20 = filt_b (bias*cut folding), rows 21..31 = 0. K=32 -> kt=0, NT=72.
__global__ void pack_wpad(const float* __restrict__ filt_W, const float* __restrict__ filt_b,
                          __bf16* __restrict__ out) {
    long i = (long)blockIdx.x * blockDim.x + threadIdx.x;
    if (i >= 32L * 1152) return;
    int  j    = (int)(i & 15);
    int  lane = (int)((i >> 4) & 31);
    long tile = i >> 9;                       // = nt, 0..71
    int  c    = (int)tile * 16 + (lane & 15);
    int  k    = ((lane >> 4) << 4) + j;
    float v = (k < 20) ? filt_W[k * 1152 + c] : ((k == 20) ? filt_b[c] : 0.f);
    out[i] = (__bf16)v;
}

__global__ void prep_edges(const float* __restrict__ edge_vec,
                           __bf16* __restrict__ phiB, float* __restrict__ dirF, int E_) {
    int e = blockIdx.x * blockDim.x + threadIdx.x;
    if (e >= E_) return;
    float ex = edge_vec[e * 3 + 0], ey = edge_vec[e * 3 + 1], ez = edge_vec[e * 3 + 2];
    float d = sqrtf(ex * ex + ey * ey + ez * ez + 1e-8f);
    float inv = 1.f / d;
    dirF[e * 3 + 0] = ex * inv;
    dirF[e * 3 + 1] = ey * inv;
    dirF[e * 3 + 2] = ez * inv;
    float cut = (d < 5.0f) ? 0.5f * (__cosf(PI_ * d * 0.2f) + 1.0f) : 0.f;
    const float gamma = 8.0f;   // 0.5 / (RADIUS/NRBF)^2
#pragma unroll
    for (int k = 0; k < 20; ++k) {
        float mu = 5.0f * (float)k / 19.0f;
        float dd = d - mu;
        phiB[(size_t)e * 32 + k] = (__bf16)(__expf(-gamma * dd * dd) * cut);
    }
    phiB[(size_t)e * 32 + 20] = (__bf16)cut;    // bias row: (b * cut)
#pragma unroll
    for (int k = 21; k < 32; ++k) phiB[(size_t)e * 32 + k] = (__bf16)0.f;
}

__global__ void embed_kernel(const int* __restrict__ z, const float* __restrict__ emb,
                             float* sF, __bf16* sB, float* vF, __bf16* vB, int n_nodes) {
    int i = blockIdx.x * blockDim.x + threadIdx.x;
    if (i >= n_nodes * Hdim) return;
    int n = i >> 7, h = i & 127;
    float s = emb[(size_t)z[n] * Hdim + h];
    sF[i] = s; sB[i] = (__bf16)s;
#pragma unroll
    for (int a = 0; a < 3; ++a) {
        size_t j = ((size_t)n * 3 + a) * Hdim + h;
        vF[j] = 0.f; vB[j] = (__bf16)0.f;
    }
}

__global__ void apply_msg_kernel(float* sF, float* vF, __bf16* vB,
                                 const float* __restrict__ dsAcc,
                                 const float* __restrict__ dvAcc, int n_nodes) {
    int i = blockIdx.x * blockDim.x + threadIdx.x;
    if (i >= n_nodes * Hdim) return;
    int n = i >> 7, h = i & 127;
    sF[i] = sF[i] + clipf(dsAcc[i]);
#pragma unroll
    for (int a = 0; a < 3; ++a) {
        size_t j = ((size_t)n * 3 + a) * Hdim + h;
        float vv = vF[j] + clipf(dvAcc[j]);
        vF[j] = vv; vB[j] = (__bf16)vv;
    }
}

__global__ void make_ts_kernel(const float* __restrict__ sF, const float* __restrict__ vwOut,
                               float* __restrict__ dotF, __bf16* __restrict__ tsB, int n_nodes) {
    int i = blockIdx.x * blockDim.x + threadIdx.x;
    if (i >= n_nodes * Hdim) return;
    int n = i >> 7, h = i & 127;
    float vl0 = vwOut[((size_t)n * 3 + 0) * 256 + h];
    float vl1 = vwOut[((size_t)n * 3 + 1) * 256 + h];
    float vl2 = vwOut[((size_t)n * 3 + 2) * 256 + h];
    float vr0 = vwOut[((size_t)n * 3 + 0) * 256 + 128 + h];
    float vr1 = vwOut[((size_t)n * 3 + 1) * 256 + 128 + h];
    float vr2 = vwOut[((size_t)n * 3 + 2) * 256 + 128 + h];
    dotF[i] = vl0 * vr0 + vl1 * vr1 + vl2 * vr2;
    float vn = sqrtf(vl0 * vl0 + vl1 * vl1 + vl2 * vl2 + 1e-8f);
    tsB[(size_t)n * 256 + h]       = (__bf16)sF[i];
    tsB[(size_t)n * 256 + 128 + h] = (__bf16)vn;
}

__global__ void apply_update_kernel(float* sF, __bf16* sB, float* vF,
                                    const float* __restrict__ mm,
                                    const float* __restrict__ dotF,
                                    const float* __restrict__ vwOut, int n_nodes) {
    int i = blockIdx.x * blockDim.x + threadIdx.x;
    if (i >= n_nodes * Hdim) return;
    int n = i >> 7, h = i & 127;
    float ds2 = mm[(size_t)n * H3 + h];
    float dvu = mm[(size_t)n * H3 + 128 + h];
    float dsv = mm[(size_t)n * H3 + 256 + h];
    float s = sF[i] + clipf(ds2 + dsv * dotF[i]);
    sF[i] = s; sB[i] = (__bf16)s;
#pragma unroll
    for (int a = 0; a < 3; ++a) {
        size_t j = ((size_t)n * 3 + a) * Hdim + h;
        float vr = vwOut[((size_t)n * 3 + a) * 256 + 128 + h];
        vF[j] = vF[j] + clipf(dvu * vr);
    }
}

__global__ void copy_out_kernel(const float* __restrict__ sF, const float* __restrict__ vF,
                                float* __restrict__ out, int n_nodes) {
    long i = (long)blockIdx.x * blockDim.x + threadIdx.x;
    long total = (long)n_nodes * 512;
    if (i >= total) return;
    long ns = (long)n_nodes * Hdim;
    out[i] = (i < ns) ? sF[i] : vF[i - ns];
}

// ---------------------------------------------------------------------------
extern "C" void kernel_launch(void* const* d_in, const int* in_sizes, int n_in,
                              void* d_out, int out_size, void* d_ws, size_t ws_size,
                              hipStream_t stream)
{
    const int*   z         = (const int*)d_in[0];
    const float* edge_vec  = (const float*)d_in[1];
    const int*   senders   = (const int*)d_in[2];
    const int*   receivers = (const int*)d_in[3];
    const float* emb       = (const float*)d_in[4];
    const float* filt_W    = (const float*)d_in[5];
    const float* filt_b    = (const float*)d_in[6];
    const float* iW1       = (const float*)d_in[7];
    const float* ib1       = (const float*)d_in[8];
    const float* iW2       = (const float*)d_in[9];
    const float* ib2       = (const float*)d_in[10];
    const float* mW1       = (const float*)d_in[11];
    const float* mb1       = (const float*)d_in[12];
    const float* mW2       = (const float*)d_in[13];
    const float* mb2       = (const float*)d_in[14];
    const float* vW        = (const float*)d_in[15];

    const int Nn = in_sizes[0];     // 10000
    const int Ee = in_sizes[2];     // 160000
    const int L_ = 3;

    char* ws = (char*)d_ws;
    size_t off = 0;
    auto alloc = [&](size_t bytes) -> void* {
        void* p = ws + off;
        off = (off + bytes + 255) & ~(size_t)255;
        return p;
    };
    float*  sF    = (float*)alloc((size_t)Nn * Hdim * 4);
    float*  vF    = (float*)alloc((size_t)Nn * H3 * 4);
    float*  xbuf  = (float*)alloc((size_t)Nn * H3 * 4);   // reused: x then mm
    float*  dsAcc = (float*)alloc((size_t)Nn * Hdim * 4);
    float*  dvAcc = (float*)alloc((size_t)Nn * H3 * 4);
    float*  vwOut = (float*)alloc((size_t)Nn * 3 * 256 * 4);
    float*  dotF  = (float*)alloc((size_t)Nn * Hdim * 4);
    float*  dirF  = (float*)alloc((size_t)Ee * 3 * 4);
    __bf16* sB    = (__bf16*)alloc((size_t)Nn * Hdim * 2);
    __bf16* vB    = (__bf16*)alloc((size_t)Nn * H3 * 2);
    __bf16* tsB   = (__bf16*)alloc((size_t)Nn * 256 * 2);
    __bf16* phiB  = (__bf16*)alloc((size_t)Ee * 32 * 2);
    __bf16* wPadP = (__bf16*)alloc((size_t)32 * 1152 * 2);
    __bf16* iW1p  = (__bf16*)alloc((size_t)L_ * Hdim * Hdim * 2);
    __bf16* iW2p  = (__bf16*)alloc((size_t)L_ * Hdim * H3 * 2);
    __bf16* mW1p  = (__bf16*)alloc((size_t)L_ * 256 * Hdim * 2);
    __bf16* mW2p  = (__bf16*)alloc((size_t)L_ * Hdim * H3 * 2);
    __bf16* vWp   = (__bf16*)alloc((size_t)L_ * Hdim * 256 * 2);

    const int TB = 256;
    auto g1 = [&](long n) { return dim3((unsigned)((n + TB - 1) / TB)); };

    // ---- one-time prep (deterministic, re-run every call) ----
    for (int l = 0; l < L_; ++l) {
        pack_B<<<g1((long)Hdim * Hdim), TB, 0, stream>>>(
            iW1 + (size_t)l * Hdim * Hdim, iW1p + (size_t)l * Hdim * Hdim, Hdim, (long)Hdim * Hdim);
        pack_B<<<g1((long)Hdim * H3), TB, 0, stream>>>(
            iW2 + (size_t)l * Hdim * H3, iW2p + (size_t)l * Hdim * H3, H3, (long)Hdim * H3);
        pack_B<<<g1((long)256 * Hdim), TB, 0, stream>>>(
            mW1 + (size_t)l * 256 * Hdim, mW1p + (size_t)l * 256 * Hdim, Hdim, (long)256 * Hdim);
        pack_B<<<g1((long)Hdim * H3), TB, 0, stream>>>(
            mW2 + (size_t)l * Hdim * H3, mW2p + (size_t)l * Hdim * H3, H3, (long)Hdim * H3);
        pack_B<<<g1((long)Hdim * 256), TB, 0, stream>>>(
            vW + (size_t)l * Hdim * 256, vWp + (size_t)l * Hdim * 256, 256, (long)Hdim * 256);
    }
    pack_wpad<<<g1(32L * 1152), TB, 0, stream>>>(filt_W, filt_b, wPadP);
    prep_edges<<<g1(Ee), TB, 0, stream>>>(edge_vec, phiB, dirF, Ee);
    embed_kernel<<<g1((long)Nn * Hdim), TB, 0, stream>>>(z, emb, sF, sB, vF, vB, Nn);

    for (int l = 0; l < L_; ++l) {
        zero_f32<<<g1((long)Nn * Hdim), TB, 0, stream>>>(dsAcc, (long)Nn * Hdim);
        zero_f32<<<g1((long)Nn * H3),   TB, 0, stream>>>(dvAcc, (long)Nn * H3);

        // x = silu(s @ iW1 + ib1) @ iW2 + ib2
        fused_mlp_kernel<128><<<dim3(Nn / 16), 32, 0, stream>>>(
            sB, iW1p + (size_t)l * Hdim * Hdim, ib1 + (size_t)l * Hdim,
            iW2p + (size_t)l * Hdim * H3, ib2 + (size_t)l * H3, xbuf);

        // filters (WMMA, fused) + message + scatter
        edge_kernel<<<dim3(Ee / 16), 32, 0, stream>>>(
            phiB, wPadP, xbuf, vF, dirF, senders, receivers, dsAcc, dvAcc, l);

        apply_msg_kernel<<<g1((long)Nn * Hdim), TB, 0, stream>>>(sF, vF, vB, dsAcc, dvAcc, Nn);

        // [vl | vr] = v @ vW
        gemm_vw_kernel<<<dim3(Nn * 3 / 16), 32, 0, stream>>>(
            vB, vWp + (size_t)l * Hdim * 256, vwOut);

        make_ts_kernel<<<g1((long)Nn * Hdim), TB, 0, stream>>>(sF, vwOut, dotF, tsB, Nn);

        // mm = silu(ts @ mW1 + mb1) @ mW2 + mb2
        fused_mlp_kernel<256><<<dim3(Nn / 16), 32, 0, stream>>>(
            tsB, mW1p + (size_t)l * 256 * Hdim, mb1 + (size_t)l * Hdim,
            mW2p + (size_t)l * Hdim * H3, mb2 + (size_t)l * H3, xbuf);

        apply_update_kernel<<<g1((long)Nn * Hdim), TB, 0, stream>>>(
            sF, sB, vF, xbuf, dotF, vwOut, Nn);
    }

    copy_out_kernel<<<g1((long)Nn * 512), TB, 0, stream>>>(sF, vF, (float*)d_out, Nn);
}
